// GCN_4509715661020
// MI455X (gfx1250) — matrix operand order repaired and checked by reading, not compiled, not run on hardware
//
#include <hip/hip_runtime.h>

// ---------------------------------------------------------------------------
// GCN forward for MI455X (gfx1250, wave32, WMMA).
//   W1t/W2t: one-time f32->f16 transpose of weights (L2-resident)
//   support = x@W1            : f16 WMMA fp32-accum, branchless b128 staging,
//                               B staged via global_load_async_to_lds_b128
//   h = spmm(A, support)+b1   : L2-resident atomic-f32 scatter
//   h = relu(h) -> f16
//   support2 = h@W2           : f16 WMMA, A+B staged via async LDS DMA
//   out = spmm(A, support2)+b2: atomic-f32 scatter
// ---------------------------------------------------------------------------

#define NN 50000
#define NE 800000
#define NF 512
#define NH 128
#define NC 40
#define NCP 48   // NC padded to 3 WMMA tiles

typedef __attribute__((ext_vector_type(16))) _Float16 v16h;
typedef __attribute__((ext_vector_type(8)))  _Float16 v8h;
typedef __attribute__((ext_vector_type(4)))  _Float16 v4h;
typedef __attribute__((ext_vector_type(8)))  float    v8f;
typedef __attribute__((ext_vector_type(4)))  float    v4f;

// A fragment (16x32 f16): lane holds row m=lane%16, K = [kb..kb+7] ++ [kb+16..kb+23],
// kb = (lane<16 ? 0 : 8). rowbase points at an LDS row (contiguous K).
__device__ __forceinline__ v16h frag_a(const _Float16* rowbase, int kb) {
  v8h lo = *(const v8h*)(rowbase + kb);
  v8h hi = *(const v8h*)(rowbase + kb + 16);
  return __builtin_shufflevector(lo, hi, 0,1,2,3,4,5,6,7,8,9,10,11,12,13,14,15);
}

// B fragment (32x16 f16), B transposed in LDS as Bt[n][k]:
// lane holds col n=lane%16, K = [kb..kb+7] ++ [kb+16..kb+23], kb = kbB/2 = (lane<16 ? 0 : 8). 16B aligned.
__device__ __forceinline__ v16h frag_b(const _Float16* rowbase, int kbB) {
  v8h lo = *(const v8h*)(rowbase + (kbB >> 1));
  v8h hi = *(const v8h*)(rowbase + (kbB >> 1) + 16);
  return __builtin_shufflevector(lo, hi, 0,1,2,3,4,5,6,7,8,9,10,11,12,13,14,15);
}

__device__ __forceinline__ void atom_add_f32(float* p, float v) {
  __hip_atomic_fetch_add(p, v, __ATOMIC_RELAXED, __HIP_MEMORY_SCOPE_AGENT);
}

__device__ __forceinline__ void async_lds_b128(unsigned lds_off, const void* gaddr) {
  asm volatile("global_load_async_to_lds_b128 %0, %1, off"
               :: "v"(lds_off), "v"(gaddr) : "memory");
}
__device__ __forceinline__ void wait_async0() {
  asm volatile("s_wait_asynccnt 0" ::: "memory");
}

// ---------------- weight pre-transpose / convert -----------------------------
__global__ __launch_bounds__(256) void gcn_w1t(const float* __restrict__ W1,
                                               _Float16* __restrict__ W1t) {
  int t = blockIdx.x * 256 + threadIdx.x;          // 65536 = 512*128
  int k = t >> 7, n = t & 127;                     // W1[k][n] row-major read
  W1t[(size_t)n * NF + k] = (_Float16)W1[t];
}

__global__ __launch_bounds__(256) void gcn_w2t(const float* __restrict__ W2,
                                               _Float16* __restrict__ W2t) {
  int t = blockIdx.x * 256 + threadIdx.x;          // 6144 = 48*128
  if (t >= NCP * NH) return;
  int n = t >> 7, k = t & 127;
  W2t[t] = (n < NC) ? (_Float16)W2[(size_t)k * NC + n] : (_Float16)0.0f;
}

// ---------------- GEMM1: support[NN,128] = x[NN,512] @ W1[512,128] -----------
__global__ __launch_bounds__(256) void gcn_gemm1(const float* __restrict__ X,
                                                 const _Float16* __restrict__ W1t,
                                                 float* __restrict__ C) {
  __shared__ __align__(64) _Float16 As[128 * 32];   // A tile rows x K32 (8 KB)
  __shared__ __align__(64) _Float16 Bs[128 * 32];   // Bt[n][k]        (8 KB)
  const int tid  = threadIdx.x;
  const int row0 = blockIdx.x * 128;
  const int wave = tid >> 5, lane = tid & 31;
  const int lm   = lane & 15;
  const int kbA  = (lane >> 4) << 3;
  const int kbB  = (lane >> 4) << 4;
  v8f acc[8] = {};

  for (int k0 = 0; k0 < NF; k0 += 32) {
    __syncthreads();   // all waves done reading LDS from previous step
    // stage A: 128 rows x 32 K as 1024 x (4 f32 -> 4 f16), branchless
#pragma unroll
    for (int i = 0; i < 4; ++i) {
      int c  = tid + i * 256;
      int r  = c >> 3, kc = (c & 7) * 4;
      int rr = row0 + r;
      int rc = rr < NN ? rr : NN - 1;
      v4f v = *(const v4f*)(X + (size_t)rc * NF + k0 + kc);
      if (rr >= NN) v = (v4f){0.0f, 0.0f, 0.0f, 0.0f};
      *(v4h*)(&As[r * 32 + kc]) = __builtin_convertvector(v, v4h);
    }
    // stage Bt: pure 8 KB copy from W1t via async LDS DMA (512 x 16B granules)
#pragma unroll
    for (int i = 0; i < 2; ++i) {
      int c = tid + i * 256;
      int n = c >> 2, ko = (c & 3) * 8;
      async_lds_b128((unsigned)(size_t)(&Bs[n * 32 + ko]),
                     W1t + (size_t)n * NF + k0 + ko);
    }
    wait_async0();
    __syncthreads();

    v16h a = frag_a(&As[(wave * 16 + lm) * 32], kbA);
    v16h bf[8];
#pragma unroll
    for (int nt = 0; nt < 8; ++nt)
      bf[nt] = frag_b(&Bs[(nt * 16 + lm) * 32], kbB);
#pragma unroll
    for (int nt = 0; nt < 8; ++nt)
      acc[nt] = __builtin_amdgcn_wmma_f32_16x16x32_f16(
          false, a, false, bf[nt], (short)0, acc[nt], false, false);
  }

  // Epilogue. C/D layout: VGPR j -> row j (lanes 0-15) or j+8 (lanes 16-31).
  const int half = (lane >> 4) << 3;
  const int rb   = row0 + wave * 16 + half;
  float* cp = C + (size_t)rb * NH + lm;
  if (row0 + 128 <= NN) {            // uniform fast path: 390 of 391 blocks
#pragma unroll
    for (int nt = 0; nt < 8; ++nt)
#pragma unroll
      for (int j = 0; j < 8; ++j)
        cp[(size_t)j * NH + nt * 16] = acc[nt][j];
  } else {
#pragma unroll
    for (int nt = 0; nt < 8; ++nt)
#pragma unroll
      for (int j = 0; j < 8; ++j)
        if (rb + j < NN) cp[(size_t)j * NH + nt * 16] = acc[nt][j];
  }
}

// ---------------- SpMM1 scatter: h[dst] += w * support[src], 128 feats ------
__global__ __launch_bounds__(256) void gcn_spmm1(const int* __restrict__ src,
                                                 const int* __restrict__ dst,
                                                 const float* __restrict__ w,
                                                 const float* __restrict__ sup,
                                                 float* __restrict__ h) {
  int t = blockIdx.x * 256 + threadIdx.x;
  int e = t >> 5, lane = t & 31;
  if (e >= NE) return;
  int s = src[e], d = dst[e];
  float we = w[e];
  const float* sp = sup + (size_t)s * NH;
  float* hp = h + (size_t)d * NH;
#pragma unroll
  for (int j = 0; j < 4; ++j) {
    int f = lane + j * 32;
    atom_add_f32(&hp[f], we * sp[f]);
  }
}

// ---------------- bias + ReLU + f32->f16 ------------------------------------
__global__ __launch_bounds__(256) void gcn_bias_relu(const float* __restrict__ h,
                                                     const float* __restrict__ b1,
                                                     _Float16* __restrict__ hf) {
  long i = (long)blockIdx.x * 256 + threadIdx.x;
  if (i >= (long)NN * NH) return;
  float v = h[i] + b1[(int)(i & (NH - 1))];
  hf[i] = (_Float16)(v > 0.0f ? v : 0.0f);
}

// ---------------- GEMM2: support2[NN,40] = h[NN,128] @ W2[128,40] -----------
__global__ __launch_bounds__(256) void gcn_gemm2(const _Float16* __restrict__ H,
                                                 const _Float16* __restrict__ W2t,
                                                 float* __restrict__ C) {
  __shared__ __align__(64) _Float16 As[128 * 128];  // 32 KB
  __shared__ __align__(64) _Float16 Bs[NCP * 128];  // Bt[n][k], 12 KB
  const int tid  = threadIdx.x;
  const int row0 = blockIdx.x * 128;

  // stage Bt: 12 KB async copy (768 x 16B granules, 3 per thread)
#pragma unroll
  for (int i = 0; i < 3; ++i) {
    int c = tid + i * 256;
    async_lds_b128((unsigned)(size_t)(&Bs[c * 8]), W2t + (size_t)c * 8);
  }
  // stage A: 32 KB async copy (2048 x 16B granules, 8 per thread)
#pragma unroll
  for (int i = 0; i < 8; ++i) {
    int c   = tid + i * 256;
    int r   = c >> 4;
    int off = (c & 15) * 8;
    if (row0 + r < NN) {
      async_lds_b128((unsigned)(size_t)(&As[r * 128 + off]),
                     H + (size_t)(row0 + r) * NH + off);
    } else {
      v8h z = {};
      *(v8h*)(&As[r * 128 + off]) = z;
    }
  }
  wait_async0();
  __syncthreads();

  const int wave = tid >> 5, lane = tid & 31;
  const int lm   = lane & 15;
  const int kbA  = (lane >> 4) << 3;
  const int kbB  = (lane >> 4) << 4;
  v8f acc[3] = {};
#pragma unroll
  for (int ks = 0; ks < 4; ++ks) {
    v16h a = frag_a(&As[(wave * 16 + lm) * 128 + ks * 32], kbA);
    v16h bf[3];
#pragma unroll
    for (int nt = 0; nt < 3; ++nt)
      bf[nt] = frag_b(&Bs[(nt * 16 + lm) * 128 + ks * 32], kbB);
#pragma unroll
    for (int nt = 0; nt < 3; ++nt)
      acc[nt] = __builtin_amdgcn_wmma_f32_16x16x32_f16(
          false, a, false, bf[nt], (short)0, acc[nt], false, false);
  }

  // Epilogue: cols 0..31 (nt=0,1) always valid; nt=2 valid only for lm<8.
  const int half = (lane >> 4) << 3;
  const int rb   = row0 + wave * 16 + half;
  float* cp = C + (size_t)rb * NC + lm;
  if (row0 + 128 <= NN) {            // uniform fast path
#pragma unroll
    for (int nt = 0; nt < 2; ++nt)
#pragma unroll
      for (int j = 0; j < 8; ++j)
        cp[(size_t)j * NC + nt * 16] = acc[nt][j];
    if (lm < 8) {
#pragma unroll
      for (int j = 0; j < 8; ++j)
        cp[(size_t)j * NC + 32] = acc[2][j];
    }
  } else {
#pragma unroll
    for (int nt = 0; nt < 2; ++nt)
#pragma unroll
      for (int j = 0; j < 8; ++j)
        if (rb + j < NN) cp[(size_t)j * NC + nt * 16] = acc[nt][j];
    if (lm < 8) {
#pragma unroll
      for (int j = 0; j < 8; ++j)
        if (rb + j < NN) cp[(size_t)j * NC + 32] = acc[2][j];
    }
  }
}

// ---------------- out init: out[n][c] = b2[c] --------------------------------
__global__ __launch_bounds__(256) void gcn_out_init(const float* __restrict__ b2,
                                                    float* __restrict__ out) {
  long i = (long)blockIdx.x * 256 + threadIdx.x;
  if (i >= (long)NN * NC) return;
  out[i] = b2[(int)(i % NC)];
}

// ---------------- SpMM2 scatter: out[dst] += w * support2[src], 40 feats ----
__global__ __launch_bounds__(256) void gcn_spmm2(const int* __restrict__ src,
                                                 const int* __restrict__ dst,
                                                 const float* __restrict__ w,
                                                 const float* __restrict__ sup,
                                                 float* __restrict__ out) {
  int t = blockIdx.x * 256 + threadIdx.x;
  int e = t >> 3, l8 = t & 7;
  if (e >= NE) return;
  int s = src[e], d = dst[e];
  float we = w[e];
  const float* sp = sup + (size_t)s * NC;
  float* op = out + (size_t)d * NC;
#pragma unroll
  for (int j = 0; j < 5; ++j) {
    int f = l8 + j * 8;
    atom_add_f32(&op[f], we * sp[f]);
  }
}

extern "C" void kernel_launch(void* const* d_in, const int* in_sizes, int n_in,
                              void* d_out, int out_size, void* d_ws, size_t ws_size,
                              hipStream_t stream) {
  const float* x    = (const float*)d_in[0];
  const float* W1   = (const float*)d_in[1];
  const float* b1   = (const float*)d_in[2];
  const float* W2   = (const float*)d_in[3];
  const float* b2   = (const float*)d_in[4];
  const int*   esrc = (const int*)d_in[5];
  const int*   edst = (const int*)d_in[6];
  const float* ew   = (const float*)d_in[7];
  float* out = (float*)d_out;

  char* ws = (char*)d_ws;
  float*    support  = (float*)(ws);                 // 25.6 MB
  float*    h        = (float*)(ws + 25600000);      // 25.6 MB
  _Float16* hf16     = (_Float16*)(ws + 51200000);   // 12.8 MB
  float*    support2 = (float*)(ws + 64000000);      //  8.0 MB
  _Float16* W1t      = (_Float16*)(ws + 72000256);   // 128 KB
  _Float16* W2t      = (_Float16*)(ws + 72150016);   //  12 KB

  const int mblocks = (NN + 127) / 128;              // 391

  gcn_w1t<<<(NF * NH) / 256, 256, 0, stream>>>(W1, W1t);
  gcn_w2t<<<(NCP * NH + 255) / 256, 256, 0, stream>>>(W2, W2t);
  hipMemsetAsync(h, 0, (size_t)NN * NH * sizeof(float), stream);

  gcn_gemm1<<<mblocks, 256, 0, stream>>>(x, W1t, support);
  gcn_spmm1<<<(NE * 32) / 256, 256, 0, stream>>>(esrc, edst, ew, support, h);
  gcn_bias_relu<<<((long)NN * NH + 255) / 256, 256, 0, stream>>>(h, b1, hf16);
  gcn_gemm2<<<mblocks, 256, 0, stream>>>(hf16, W2t, support2);
  gcn_out_init<<<((long)NN * NC + 255) / 256, 256, 0, stream>>>(b2, out);
  gcn_spmm2<<<(NE * 8) / 256, 256, 0, stream>>>(esrc, edst, ew, support2, out);
}
